// VectorQuantizer_25984552141284
// MI455X (gfx1250) — matrix operand order repaired and checked by reading, not compiled, not run on hardware
//
#include <hip/hip_runtime.h>

// ---------------------------------------------------------------------------
// VQ-VAE vector quantizer for MI455X (gfx1250, wave32, WMMA).
// dist(n,k) = ||z_n||^2 + ||e_k||^2 - 2 z_n.e_k ; argmin_k drops ||z_n||^2.
// GEMM z.e^T in split-bf16 (hi+lo) with v_wmma_f32_16x16x32_bf16.
// B tiles staged to LDS once per 4-wave workgroup with
// global_load_async_to_lds_b128 (ASYNCcnt), double-buffered against compute.
// ---------------------------------------------------------------------------

#define VQ_N      16384     // 16*32*32 points
#define VQ_D      256       // embed dim
#define VQ_K      8192      // codebook size
#define VQ_SPLIT  8         // codebook splits per row-group
#define VQ_CPS    (VQ_K / VQ_SPLIT)   // 1024 codes per split
#define VQ_TILES  (VQ_CPS / 16)       // 64 code tiles of 16
#define VQ_DCH    (VQ_D / 32)         // 8 depth chunks of 32

typedef __attribute__((ext_vector_type(16))) __bf16 v16bf;
typedef __attribute__((ext_vector_type(8)))  float  v8f;

union V16U { v16bf v; uint4 u4[2]; };

__device__ __forceinline__ unsigned short f2bf(float f) {
  unsigned int u = __float_as_uint(f);
  u += 0x7fffu + ((u >> 16) & 1u);        // round-to-nearest-even
  return (unsigned short)(u >> 16);
}
__device__ __forceinline__ float bf2f(unsigned short h) {
  return __uint_as_float(((unsigned int)h) << 16);
}
__device__ __forceinline__ v16bf load16bf(const unsigned short* p) {
  V16U t;
  t.u4[0] = *reinterpret_cast<const uint4*>(p);
  t.u4[1] = *reinterpret_cast<const uint4*>(p + 8);
  return t.v;
}
__device__ __forceinline__ v8f wmma_bf16(v16bf a, v16bf b, v8f c) {
  return __builtin_amdgcn_wmma_f32_16x16x32_bf16(false, a, false, b,
                                                 (short)0, c, false, false);
}

// Async global->LDS 16-byte copy (per-lane addresses, tracked by ASYNCcnt).
// Low 32 bits of a flat shared pointer are the LDS byte address (ISA 10.2).
__device__ __forceinline__ void async_copy_b128(unsigned short* lds_p,
                                                const unsigned short* g_p) {
  unsigned int lds_addr = (unsigned int)(uintptr_t)lds_p;
  asm volatile("global_load_async_to_lds_b128 %0, %1, off"
               :: "v"(lds_addr), "v"(g_p) : "memory");
}
__device__ __forceinline__ void wait_async0() {
  asm volatile("s_wait_asynccnt 0x0" ::: "memory");
}

// --- prep: z [16,256,32,32] -> z_hi/z_lo [N,256] bf16 (transposed) ----------
__global__ void vq_prep_z(const float* __restrict__ z,
                          unsigned short* __restrict__ z_hi,
                          unsigned short* __restrict__ z_lo) {
  int t   = blockIdx.x * 256 + threadIdx.x;   // flat over input [b][c][h][w]
  int b   = t >> 18;
  int rem = t & 262143;
  int c   = rem >> 10;
  int hw  = rem & 1023;
  float v = z[t];
  unsigned short h = f2bf(v);
  unsigned short l = f2bf(v - bf2f(h));
  int n = (b << 10) + hw;                     // row in [N,256]
  z_hi[n * VQ_D + c] = h;
  z_lo[n * VQ_D + c] = l;
}

// --- prep: embedding fp32 -> bf16 hi/lo -------------------------------------
__global__ void vq_prep_e(const float* __restrict__ e,
                          unsigned short* __restrict__ e_hi,
                          unsigned short* __restrict__ e_lo) {
  int t = blockIdx.x * 256 + threadIdx.x;     // over K*D
  float v = e[t];
  unsigned short h = f2bf(v);
  e_hi[t] = h;
  e_lo[t] = f2bf(v - bf2f(h));
}

// --- ||e_k||^2 --------------------------------------------------------------
__global__ void vq_enorm(const float* __restrict__ e, float* __restrict__ en) {
  __shared__ float s[256];
  int k = blockIdx.x, c = threadIdx.x;
  float v = e[k * VQ_D + c];
  s[c] = v * v;
  __syncthreads();
  for (int o = 128; o > 0; o >>= 1) {
    if (c < o) s[c] += s[c + o];
    __syncthreads();
  }
  if (c == 0) en[k] = s[0];
}

// --- main: split-bf16 GEMM + running argmin ---------------------------------
// Block = 128 threads (4 waves x 32 rows = 128 rows). Grid = (N/128)*VQ_SPLIT.
// B tiles staged once per block into LDS (async), shared by all 4 waves.
__global__ void __launch_bounds__(128)
vq_gemm_argmin(const unsigned short* __restrict__ z_hi,
               const unsigned short* __restrict__ z_lo,
               const unsigned short* __restrict__ e_hi,
               const unsigned short* __restrict__ e_lo,
               const float* __restrict__ e_norms,
               float* __restrict__ pval, int* __restrict__ pidx) {
  __shared__ __align__(16) unsigned short lds_hi[2][16 * VQ_D];  // 2 x 8 KB
  __shared__ __align__(16) unsigned short lds_lo[2][16 * VQ_D];  // 2 x 8 KB

  const int tid     = threadIdx.x;                // 0..127
  const int lane    = tid & 31;                   // lane within wave
  const int wave    = tid >> 5;                   // 0..3
  const int half    = (lane >= 16) ? 1 : 0;       // K-half selector
  const int l15     = lane & 15;
  const int rg      = blockIdx.x / VQ_SPLIT;
  const int sp      = blockIdx.x % VQ_SPLIT;
  const int rowBase = rg * 128 + wave * 32;       // 32 rows per wave
  const int cBase   = sp * VQ_CPS;

  // Resident A tiles: 2 row-tiles x 8 d-chunks, hi & lo.
  // A k order used here (contiguous, not ISA table order): lanes 0-15 hold K 0..15, lanes 16-31 K 16..31 (matches B).
  v16bf a_hi[2][VQ_DCH], a_lo[2][VQ_DCH];
#pragma unroll
  for (int t = 0; t < 2; ++t) {
#pragma unroll
    for (int dc = 0; dc < VQ_DCH; ++dc) {
      size_t base = (size_t)(rowBase + t * 16 + l15) * VQ_D + dc * 32 + half * 16;
      V16U th, tl;
      th.u4[0] = *reinterpret_cast<const uint4*>(z_hi + base);
      th.u4[1] = *reinterpret_cast<const uint4*>(z_hi + base + 8);
      tl.u4[0] = *reinterpret_cast<const uint4*>(z_lo + base);
      tl.u4[1] = *reinterpret_cast<const uint4*>(z_lo + base + 8);
      a_hi[t][dc] = th.v;
      a_lo[t][dc] = tl.v;
    }
  }

  float mval[2][8];
  int   midx[2][8];
#pragma unroll
  for (int t = 0; t < 2; ++t)
#pragma unroll
    for (int j = 0; j < 8; ++j) { mval[t][j] = 3.4e38f; midx[t][j] = 0; }

  // Stage a 16-code B tile (hi+lo, 16 KB) into one LDS buffer.
  // 128 threads x 16 B x 4 iters per precision. Tile rows are contiguous
  // in the row-major codebook, so this is a flat 8 KB copy per precision.
  auto stage = [&](int c0, int buf) {
    const unsigned short* gh = e_hi + (size_t)c0 * VQ_D;
    const unsigned short* gl = e_lo + (size_t)c0 * VQ_D;
#pragma unroll
    for (int i = 0; i < 4; ++i) {
      int eo = (i * 128 + tid) * 8;               // element offset (16B chunks)
      async_copy_b128(&lds_hi[buf][eo], gh + eo);
      async_copy_b128(&lds_lo[buf][eo], gl + eo);
    }
  };

  stage(cBase, 0);
  wait_async0();
  __syncthreads();

  for (int tile = 0; tile < VQ_TILES; ++tile) {
    const int buf = tile & 1;
    if (tile + 1 < VQ_TILES) stage(cBase + (tile + 1) * 16, buf ^ 1);

    const int c0 = cBase + tile * 16;
    const float en = e_norms[c0 + l15];

    // B k order used here (32x16 bf16, contiguous, matches A): lane n holds column (lane&15), K 0..15 (lanes<16)
    // or K 16..31 (lanes>=16) -> 32 contiguous bytes of the staged rows.
    const unsigned short* lb_h = &lds_hi[buf][l15 * VQ_D + half * 16];
    const unsigned short* lb_l = &lds_lo[buf][l15 * VQ_D + half * 16];

    v8f acc0 = {0.f, 0.f, 0.f, 0.f, 0.f, 0.f, 0.f, 0.f};
    v8f acc1 = {0.f, 0.f, 0.f, 0.f, 0.f, 0.f, 0.f, 0.f};
#pragma unroll
    for (int dc = 0; dc < VQ_DCH; ++dc) {
      v16bf bh = load16bf(lb_h + dc * 32);
      v16bf bl = load16bf(lb_l + dc * 32);
      // dot ~= zh.eh + zl.eh + zh.el  (split-bf16, near-fp32 accuracy)
      acc0 = wmma_bf16(a_hi[0][dc], bh, acc0);
      acc0 = wmma_bf16(a_lo[0][dc], bh, acc0);
      acc0 = wmma_bf16(a_hi[0][dc], bl, acc0);
      acc1 = wmma_bf16(a_hi[1][dc], bh, acc1);
      acc1 = wmma_bf16(a_lo[1][dc], bh, acc1);
      acc1 = wmma_bf16(a_hi[1][dc], bl, acc1);
    }
#pragma unroll
    for (int j = 0; j < 8; ++j) {
      float d0 = en - 2.0f * acc0[j];
      float d1 = en - 2.0f * acc1[j];
      if (d0 < mval[0][j]) { mval[0][j] = d0; midx[0][j] = c0 + l15; }
      if (d1 < mval[1][j]) { mval[1][j] = d1; midx[1][j] = c0 + l15; }
    }

    // Tile t+1 staged & everyone done reading buf before it is re-staged.
    wait_async0();
    __syncthreads();
  }

  // Reduce the 16 column candidates of each row (lanes sharing a 16-lane
  // group hold the same rows; C layout: row = j + 8*half, col = lane&15).
#pragma unroll
  for (int t = 0; t < 2; ++t) {
#pragma unroll
    for (int j = 0; j < 8; ++j) {
      float v = mval[t][j];
      int   ix = midx[t][j];
      for (int m = 8; m >= 1; m >>= 1) {
        float ov = __shfl_xor(v, m, 16);
        int   oi = __shfl_xor(ix, m, 16);
        if (ov < v || (ov == v && oi < ix)) { v = ov; ix = oi; }
      }
      if (l15 == 0) {
        int row = rowBase + t * 16 + j + half * 8;
        pval[row * VQ_SPLIT + sp] = v;
        pidx[row * VQ_SPLIT + sp] = ix;
      }
    }
  }
}

// --- merge codebook splits --------------------------------------------------
__global__ void vq_merge(const float* __restrict__ pval,
                         const int* __restrict__ pidx,
                         int* __restrict__ ids, float* __restrict__ ids_f) {
  int n = blockIdx.x * 256 + threadIdx.x;
  float best = pval[n * VQ_SPLIT];
  int   bi   = pidx[n * VQ_SPLIT];
  for (int s = 1; s < VQ_SPLIT; ++s) {
    float v = pval[n * VQ_SPLIT + s];
    int   i = pidx[n * VQ_SPLIT + s];
    if (v < best || (v == best && i < bi)) { best = v; bi = i; }
  }
  ids[n]   = bi;
  ids_f[n] = (float)bi;   // output dtype is float
}

// --- gather z_q into [b,c,h,w], per-point squared-error sums ---------------
__global__ void vq_gather_loss(const float* __restrict__ z,
                               const float* __restrict__ emb,
                               const int* __restrict__ ids,
                               float* __restrict__ zq_out,
                               float* __restrict__ bsum) {
  __shared__ float s[256];
  int n = blockIdx.x, c = threadIdx.x;
  int b = n >> 10, hw = n & 1023;
  int id = ids[n];
  float ev = emb[id * VQ_D + c];
  int io = (b << 18) + (c << 10) + hw;   // [b][c][h][w] flat
  float zv = z[io];
  zq_out[io] = ev;                       // straight-through value == z_q
  float d = ev - zv;
  s[c] = d * d;
  __syncthreads();
  for (int o = 128; o > 0; o >>= 1) {
    if (c < o) s[c] += s[c + o];
    __syncthreads();
  }
  if (c == 0) bsum[n] = s[0];
}

// --- deterministic final reduction + scalars --------------------------------
__global__ void vq_finalize(const float* __restrict__ bsum,
                            float* __restrict__ scalars) {
  __shared__ float s[256];
  float acc = 0.f;
  for (int i = threadIdx.x; i < VQ_N; i += 256) acc += bsum[i];
  s[threadIdx.x] = acc;
  __syncthreads();
  for (int o = 128; o > 0; o >>= 1) {
    if (threadIdx.x < o) s[threadIdx.x] += s[threadIdx.x + o];
    __syncthreads();
  }
  if (threadIdx.x == 0) {
    float mse = s[0] / (float)(VQ_N * VQ_D);
    scalars[0] = 1.25f * mse;   // loss
    scalars[1] = 0.25f * mse;   // commitment
    scalars[2] = mse;           // codebook
  }
}

extern "C" void kernel_launch(void* const* d_in, const int* in_sizes, int n_in,
                              void* d_out, int out_size, void* d_ws, size_t ws_size,
                              hipStream_t stream) {
  const float* z   = (const float*)d_in[0];   // [16,256,32,32]
  const float* emb = (const float*)d_in[1];   // [8192,256]
  float* out = (float*)d_out;                 // zq | loss | commit | codebook | ids

  char* ws = (char*)d_ws;
  const size_t OFF_EHI   = 0;                                  // 4 MB
  const size_t OFF_ELO   = (size_t)4  << 20;                   // 4 MB
  const size_t OFF_ZHI   = (size_t)8  << 20;                   // 8 MB
  const size_t OFF_ZLO   = (size_t)16 << 20;                   // 8 MB
  const size_t OFF_ENORM = (size_t)24 << 20;                   // 32 KB
  const size_t OFF_PVAL  = OFF_ENORM + ((size_t)64 << 10);     // 512 KB
  const size_t OFF_PIDX  = OFF_PVAL + ((size_t)512 << 10);     // 512 KB
  const size_t OFF_IDS   = OFF_PIDX + ((size_t)512 << 10);     // 64 KB
  const size_t OFF_BSUM  = OFF_IDS + ((size_t)64 << 10);       // 64 KB

  unsigned short* e_hi = (unsigned short*)(ws + OFF_EHI);
  unsigned short* e_lo = (unsigned short*)(ws + OFF_ELO);
  unsigned short* z_hi = (unsigned short*)(ws + OFF_ZHI);
  unsigned short* z_lo = (unsigned short*)(ws + OFF_ZLO);
  float* e_norms = (float*)(ws + OFF_ENORM);
  float* pval    = (float*)(ws + OFF_PVAL);
  int*   pidx    = (int*)(ws + OFF_PIDX);
  int*   ids     = (int*)(ws + OFF_IDS);
  float* bsum    = (float*)(ws + OFF_BSUM);

  float* out_scalars = out + (size_t)VQ_N * VQ_D;          // 4194304
  float* out_ids     = out_scalars + 3;                    // 4194307

  vq_prep_z<<<VQ_N * VQ_D / 256, 256, 0, stream>>>(z, z_hi, z_lo);
  vq_prep_e<<<VQ_K * VQ_D / 256, 256, 0, stream>>>(emb, e_hi, e_lo);
  vq_enorm<<<VQ_K, 256, 0, stream>>>(emb, e_norms);
  vq_gemm_argmin<<<(VQ_N / 128) * VQ_SPLIT, 128, 0, stream>>>(
      z_hi, z_lo, e_hi, e_lo, e_norms, pval, pidx);
  vq_merge<<<VQ_N / 256, 256, 0, stream>>>(pval, pidx, ids, out_ids);
  vq_gather_loss<<<VQ_N, 256, 0, stream>>>(z, emb, ids, out, bsum);
  vq_finalize<<<1, 256, 0, stream>>>(bsum, out_scalars);
}
